// AttentionMask_13048110645633
// MI455X (gfx1250) — matrix operand order repaired and checked by reading, not compiled, not run on hardware
//
#include <hip/hip_runtime.h>
#include <hip/hip_bf16.h>

typedef __attribute__((ext_vector_type(16))) _Float16     v16h;
typedef __attribute__((ext_vector_type(8)))  float        v8f;
typedef __attribute__((ext_vector_type(8)))  unsigned int v8u;

#define LOG2E 1.44269504088896340736f

union H16U8 { v16h h; v8u u; };

// B = 4, J = 32*64*64 = 131072, K = 512
// 4096 blocks x 256 threads; 1024 blocks per batch; each block = 128 rows (8 waves x 16 rows)
__global__ __launch_bounds__(256)
void attn_gate_wmma_kernel(const float* __restrict__ key,
                           const float* __restrict__ query,
                           float* __restrict__ out) {
    constexpr int J = 131072;
    constexpr int K = 512;

    __shared__ __align__(16) float     qs[K];   // q in f32 (A-tile exps)
    __shared__ __align__(32) _Float16  qh[K];   // q in f16 (B-tile column 1)
    __shared__ float wmx[8], wmn[8];            // per-wave max/min partials
    __shared__ float sctx[8 * 32];              // per-wave: 16 rows x {sumE, sumQE}

    const int tid   = threadIdx.x;
    const int lane  = tid & 31;
    const int wave  = tid >> 5;
    const int batch = blockIdx.x >> 10;         // 1024 blocks per batch
    const int brow  = (blockIdx.x & 1023) * 128;

    // ---- stage q into LDS (f32 + f16), block-reduce qmax/qmin ------------
    const float* qb = query + (size_t)batch * K;
    float v0 = qb[tid];
    float v1 = qb[tid + 256];
    qs[tid]       = v0;
    qs[tid + 256] = v1;
    qh[tid]       = (_Float16)v0;
    qh[tid + 256] = (_Float16)v1;

    float mx = fmaxf(v0, v1);
    float mn = fminf(v0, v1);
    #pragma unroll
    for (int off = 16; off > 0; off >>= 1) {
        mx = fmaxf(mx, __shfl_xor(mx, off, 32));
        mn = fminf(mn, __shfl_xor(mn, off, 32));
    }
    if (lane == 0) { wmx[wave] = mx; wmn[wave] = mn; }
    __syncthreads();

    float qmax = wmx[0], qmin = wmn[0];
    #pragma unroll
    for (int i = 1; i < 8; ++i) {
        qmax = fmaxf(qmax, wmx[i]);
        qmin = fminf(qmin, wmn[i]);
    }

    // ---- per-wave tile: 16 rows of j ------------------------------------
    const int m  = lane & 15;   // row within tile (A) / column N (B,C)
    const int hi = lane >> 4;   // lane half selects K sub-group
    const int j0 = brow + wave * 16;

    const size_t base = (size_t)batch * J + j0;
    const float x  = key[base + m];              // rows duplicated across halves
    const float xl = x * LOG2E;                  // fold log2(e) into the fma
    const float m2 = (x >= 0.0f) ? xl * qmax : xl * qmin;  // log2 of row max

    // B-tile select masks (branchless): col0 = ones, col1 = q, rest = 0
    const unsigned int sel_q    = (m == 1) ? 0xFFFFFFFFu : 0u;
    const unsigned int ones_sel = (m == 0) ? 0x3C003C00u : 0u;  // packed f16 1.0

    const float4* qs4 = reinterpret_cast<const float4*>(qs);

    v8f c = {};   // C/D accumulator: col0 = sumE, col1 = sumQE (per row)

    #pragma unroll 4
    for (int t = 0; t < K / 32; ++t) {
        const int k0 = t * 32;

        // A tile 16x32 f16: lane holds K = k0+hi*16+0..15
        float4 qa0 = qs4[(k0 + hi * 16) / 4];
        float4 qa1 = qs4[(k0 + hi * 16) / 4 + 1];
        float4 qb0 = qs4[(k0 + hi * 16 + 8) / 4];
        float4 qb1 = qs4[(k0 + hi * 16 + 8) / 4 + 1];

        v16h a;
        a[0]  = (_Float16)__builtin_amdgcn_exp2f(fmaf(xl, qa0.x, -m2));
        a[1]  = (_Float16)__builtin_amdgcn_exp2f(fmaf(xl, qa0.y, -m2));
        a[2]  = (_Float16)__builtin_amdgcn_exp2f(fmaf(xl, qa0.z, -m2));
        a[3]  = (_Float16)__builtin_amdgcn_exp2f(fmaf(xl, qa0.w, -m2));
        a[4]  = (_Float16)__builtin_amdgcn_exp2f(fmaf(xl, qa1.x, -m2));
        a[5]  = (_Float16)__builtin_amdgcn_exp2f(fmaf(xl, qa1.y, -m2));
        a[6]  = (_Float16)__builtin_amdgcn_exp2f(fmaf(xl, qa1.z, -m2));
        a[7]  = (_Float16)__builtin_amdgcn_exp2f(fmaf(xl, qa1.w, -m2));
        a[8]  = (_Float16)__builtin_amdgcn_exp2f(fmaf(xl, qb0.x, -m2));
        a[9]  = (_Float16)__builtin_amdgcn_exp2f(fmaf(xl, qb0.y, -m2));
        a[10] = (_Float16)__builtin_amdgcn_exp2f(fmaf(xl, qb0.z, -m2));
        a[11] = (_Float16)__builtin_amdgcn_exp2f(fmaf(xl, qb0.w, -m2));
        a[12] = (_Float16)__builtin_amdgcn_exp2f(fmaf(xl, qb1.x, -m2));
        a[13] = (_Float16)__builtin_amdgcn_exp2f(fmaf(xl, qb1.y, -m2));
        a[14] = (_Float16)__builtin_amdgcn_exp2f(fmaf(xl, qb1.z, -m2));
        a[15] = (_Float16)__builtin_amdgcn_exp2f(fmaf(xl, qb1.w, -m2));

        // B tile 32x16 f16 (branchless): lane holds col n=m, K = k0+hi*16+0..15
        H16U8 qv, bm;
        qv.h = *(const v16h*)(qh + k0 + hi * 16);   // 32B aligned -> 2x ds_load_b128
        #pragma unroll
        for (int i = 0; i < 8; ++i)
            bm.u[i] = (qv.u[i] & sel_q) | ones_sel;

        // D = A x B + C   (emits v_wmma_f32_16x16x32_f16)
        c = __builtin_amdgcn_wmma_f32_16x16x32_f16(
                /*neg_a=*/false, a, /*neg_b=*/false, bm.h,
                /*c_mod=*/(short)0, c, /*reuse_a=*/false, /*reuse_b=*/false);
    }

    // ---- spill D columns 0,1 through LDS (same-wave DS is in-order) ------
    if (m < 2) {
        #pragma unroll
        for (int r = 0; r < 8; ++r) {
            int row = r + 8 * hi;                 // C/D layout: M = r + 8*(lane/16)
            sctx[wave * 32 + row * 2 + m] = c[r];
        }
    }

    if (lane < 16) {   // lane == row; x already holds this row's key value
        float se  = sctx[wave * 32 + lane * 2 + 0];
        float sq  = sctx[wave * 32 + lane * 2 + 1];
        float ctx = sq / se;
        float gate = 1.0f / (1.0f + __builtin_amdgcn_exp2f(-ctx * LOG2E));
        out[base + lane] = x * gate;
    }
}

extern "C" void kernel_launch(void* const* d_in, const int* in_sizes, int n_in,
                              void* d_out, int out_size, void* d_ws, size_t ws_size,
                              hipStream_t stream) {
    const float* key   = (const float*)d_in[0];   // (4, 32, 64, 64) f32
    const float* query = (const float*)d_in[1];   // (4, 512)        f32
    float*       out   = (float*)d_out;           // (4, 32, 64, 64) f32

    // 4 batches * 1024 blocks each; 256 threads (8 waves, one 16-row tile per wave)
    attn_gate_wmma_kernel<<<4096, 256, 0, stream>>>(key, query, out);
}
